// TransformerSpatialEncoder_3728031613486
// MI455X (gfx1250) — matrix operand order repaired and checked by reading, not compiled, not run on hardware
//
#include <hip/hip_runtime.h>
#include <math.h>

// ---------------------------------------------------------------------------
// Types
// ---------------------------------------------------------------------------
typedef __bf16 bf16_t;
typedef __attribute__((ext_vector_type(8)))  __bf16 v8bf;
typedef __attribute__((ext_vector_type(16))) __bf16 v16bf;
typedef __attribute__((ext_vector_type(8)))  float  v8f;

#define LDK 40            // LDS row stride (elements); 80B, multiple of 16B
#define AS_ELTS (128*LDK) // one A buffer
#define BS_ELTS (64*LDK)  // one B buffer

// ---------------------------------------------------------------------------
// bf16 WMMA GEMM:  C[M,N] (+bias, +act) = A[M,K] @ B^T   (B stored [N][K])
//   Block tile 128x64, wave tile 32x32 (4 WMMAs / K-step), BK=32,
//   double-buffered LDS, one __syncthreads per K-step.
//   A: bf16 row-major, lda.   B: bf16 N-major ([N][K] rows), ldb.
//   Writes f32 C and/or bf16 Cb.  cb_vt==1 stores Cb V-transposed:
//   idx = (row>>9)*393216 + col*512 + (row&511)   (B=16,L=512,D=768 specific).
//   Batched via blockIdx.z = (zb, zh), zh in [0, bat_h).
//   Requires M%128==0, N%64==0, K%32==0 (true for every call here).
// ---------------------------------------------------------------------------
__global__ __launch_bounds__(256)
void gemm_bf16_wmma_kernel(const bf16_t* __restrict__ A,
                           const bf16_t* __restrict__ Bm,
                           float* __restrict__ C,
                           bf16_t* __restrict__ Cb,
                           const float* __restrict__ bias,
                           int M, int N, int K,
                           int lda, int ldb, int ldc,
                           int bat_h,
                           long long sA_b, long long sA_h,
                           long long sB_b, long long sB_h,
                           long long sC_b, long long sC_h,
                           int cb_vt, int act)
{
    __shared__ __align__(16) bf16_t As[2 * AS_ELTS];
    __shared__ __align__(16) bf16_t Bs[2 * BS_ELTS];

    const int z  = blockIdx.z;
    const int zb = z / bat_h;
    const int zh = z - zb * bat_h;
    A  += zb * sA_b + zh * sA_h;
    Bm += zb * sB_b + zh * sB_h;
    const long long coff = zb * sC_b + zh * sC_h;

    const int m0   = blockIdx.y * 128;
    const int n0   = blockIdx.x * 64;
    const int tid  = threadIdx.x;
    const int lane = tid & 31;
    const int wave = tid >> 5;
    const int wr   = wave >> 1;        // 0..3 : 32-row tile
    const int wc   = wave & 1;         // 0..1 : 32-col tile
    const int lane15 = lane & 15;
    const int kb   = (lane >> 4) * 8;  // A-frag K base within halves
    const int kb2  = (lane >> 4) * 16; // unused; B-frag uses kb

    // staging coordinates
    const int rowA = tid >> 1;          // 0..127
    const int kcA  = (tid & 1) * 16;    // 0 or 16
    const int rowB = tid >> 2;          // 0..63
    const int kcB  = (tid & 3) * 8;     // 0..24

    v8f acc[2][2] = {{{}, {}}, {{}, {}}};

    // ---- prologue: stage tile k0=0 into buffer 0 ----
    {
        v8bf a0 = *(const v8bf*)(A + (long long)(m0 + rowA) * lda + kcA);
        v8bf a1 = *(const v8bf*)(A + (long long)(m0 + rowA) * lda + kcA + 8);
        v8bf b0 = *(const v8bf*)(Bm + (long long)(n0 + rowB) * ldb + kcB);
        *(v8bf*)(&As[rowA * LDK + kcA])     = a0;
        *(v8bf*)(&As[rowA * LDK + kcA + 8]) = a1;
        *(v8bf*)(&Bs[rowB * LDK + kcB])     = b0;
    }
    __syncthreads();

    for (int k0 = 0, it = 0; k0 < K; k0 += 32, ++it) {
        const int cur = it & 1;
        const bf16_t* Ab = As + cur * AS_ELTS;
        const bf16_t* Bb = Bs + cur * BS_ELTS;
        const bool has_next = (k0 + 32) < K;

        // issue next-tile global loads early (hidden under the WMMAs)
        v8bf na0, na1, nb0;
        if (has_next) {
            na0 = *(const v8bf*)(A + (long long)(m0 + rowA) * lda + k0 + 32 + kcA);
            na1 = *(const v8bf*)(A + (long long)(m0 + rowA) * lda + k0 + 32 + kcA + 8);
            nb0 = *(const v8bf*)(Bm + (long long)(n0 + rowB) * ldb + k0 + 32 + kcB);
        }

        // ---- fragments ----
        v16bf afrag[2], bfrag[2];
#pragma unroll
        for (int mr = 0; mr < 2; ++mr) {
            const int row = wr * 32 + mr * 16 + lane15;
            v8bf lo = *(const v8bf*)(&Ab[row * LDK + kb]);
            v8bf hi = *(const v8bf*)(&Ab[row * LDK + kb + 16]);
#pragma unroll
            for (int j = 0; j < 8; ++j) { afrag[mr][j] = lo[j]; afrag[mr][j + 8] = hi[j]; }
        }
#pragma unroll
        for (int nc = 0; nc < 2; ++nc) {
            const int nn = wc * 32 + nc * 16 + lane15;
            v8bf lo = *(const v8bf*)(&Bb[nn * LDK + kb]);
            v8bf hi = *(const v8bf*)(&Bb[nn * LDK + kb + 16]);
#pragma unroll
            for (int j = 0; j < 8; ++j) { bfrag[nc][j] = lo[j]; bfrag[nc][j + 8] = hi[j]; }
        }

        acc[0][0] = __builtin_amdgcn_wmma_f32_16x16x32_bf16(false, afrag[0], false, bfrag[0],
                                                            (short)0, acc[0][0], false, false);
        acc[0][1] = __builtin_amdgcn_wmma_f32_16x16x32_bf16(false, afrag[0], false, bfrag[1],
                                                            (short)0, acc[0][1], false, false);
        acc[1][0] = __builtin_amdgcn_wmma_f32_16x16x32_bf16(false, afrag[1], false, bfrag[0],
                                                            (short)0, acc[1][0], false, false);
        acc[1][1] = __builtin_amdgcn_wmma_f32_16x16x32_bf16(false, afrag[1], false, bfrag[1],
                                                            (short)0, acc[1][1], false, false);

        // stage next tile into the other buffer
        if (has_next) {
            bf16_t* An = As + (cur ^ 1) * AS_ELTS;
            bf16_t* Bn = Bs + (cur ^ 1) * BS_ELTS;
            *(v8bf*)(&An[rowA * LDK + kcA])     = na0;
            *(v8bf*)(&An[rowA * LDK + kcA + 8]) = na1;
            *(v8bf*)(&Bn[rowB * LDK + kcB])     = nb0;
        }
        __syncthreads();
    }

    // ---- epilogue: bias + activation + store ----
#pragma unroll
    for (int mr = 0; mr < 2; ++mr) {
        const int rbase = m0 + wr * 32 + mr * 16 + (lane >> 4) * 8;
#pragma unroll
        for (int nc = 0; nc < 2; ++nc) {
            const int col = n0 + wc * 32 + nc * 16 + lane15;
            const float bv = bias ? bias[col] : 0.0f;
#pragma unroll
            for (int r = 0; r < 8; ++r) {
                float v = acc[mr][nc][r] + bv;
                if (act == 1)  // exact GELU
                    v = 0.5f * v * (1.0f + erff(v * 0.7071067811865476f));
                const int row = rbase + r;
                if (C)  C[coff + (long long)row * ldc + col] = v;
                if (Cb) {
                    long long idx;
                    if (cb_vt)  // V^T store: (B,H,DK,L) layout, L=512, D=768
                        idx = (long long)(row >> 9) * 393216 +
                              (long long)col * 512 + (row & 511);
                    else
                        idx = coff + (long long)row * ldc + col;
                    Cb[idx] = (bf16_t)v;
                }
            }
        }
    }
}

// ---------------------------------------------------------------------------
// Block reduction helpers (blockDim.x == 256)
// ---------------------------------------------------------------------------
__device__ inline float block_max(float v, float* s) {
    const int t = threadIdx.x;
    s[t] = v; __syncthreads();
    for (int o = 128; o > 0; o >>= 1) {
        if (t < o) s[t] = fmaxf(s[t], s[t + o]);
        __syncthreads();
    }
    float r = s[0]; __syncthreads(); return r;
}
__device__ inline float block_sum(float v, float* s) {
    const int t = threadIdx.x;
    s[t] = v; __syncthreads();
    for (int o = 128; o > 0; o >>= 1) {
        if (t < o) s[t] += s[t + o];
        __syncthreads();
    }
    float r = s[0]; __syncthreads(); return r;
}
__device__ inline void block_sum2(float& a, float& b, float* s1, float* s2) {
    const int t = threadIdx.x;
    s1[t] = a; s2[t] = b; __syncthreads();
    for (int o = 128; o > 0; o >>= 1) {
        if (t < o) { s1[t] += s1[t + o]; s2[t] += s2[t + o]; }
        __syncthreads();
    }
    a = s1[0]; b = s2[0]; __syncthreads();
}

// ---------------------------------------------------------------------------
// Tiled transpose + f32->bf16:  dst[N][K] = (bf16) src[K][N], per layer slice.
// grid: (N/32, K/32, n_layers); block 256 = 32x8.
// ---------------------------------------------------------------------------
__global__ __launch_bounds__(256)
void transpose_bf16_kernel(const float* __restrict__ src,
                           bf16_t* __restrict__ dst, int K, int N) {
    __shared__ float tile[32][33];
    const long long slice = (long long)blockIdx.z * K * N;
    src += slice; dst += slice;
    const int tx = threadIdx.x & 31, ty = threadIdx.x >> 5;
    const int bx = blockIdx.x * 32, by = blockIdx.y * 32;
#pragma unroll
    for (int j = 0; j < 4; ++j) {
        const int r = by + ty + j * 8;          // k index
        tile[ty + j * 8][tx] = src[(long long)r * N + bx + tx];
    }
    __syncthreads();
#pragma unroll
    for (int j = 0; j < 4; ++j) {
        const int n = bx + ty + j * 8;
        const int k = by + tx;
        dst[(long long)n * K + k] = (bf16_t)tile[tx][ty + j * 8];
    }
}

// ---------------------------------------------------------------------------
// x = src + qpos ; xb = bf16(x)
// ---------------------------------------------------------------------------
__global__ void addpos_kernel(const float* __restrict__ src,
                              const float* __restrict__ qpos,
                              float* __restrict__ x, bf16_t* __restrict__ xb,
                              long long n) {
    for (long long i = blockIdx.x * (long long)blockDim.x + threadIdx.x;
         i < n; i += (long long)gridDim.x * blockDim.x) {
        const float v = src[i] + qpos[i];
        x[i]  = v;
        xb[i] = (bf16_t)v;
    }
}

// ---------------------------------------------------------------------------
// query_pos = LayerNorm(obj_locs @ loc_w + loc_b).  One block per row (D=768).
// ---------------------------------------------------------------------------
__global__ __launch_bounds__(256)
void qpos_kernel(const float* __restrict__ locs, const float* __restrict__ W,
                 const float* __restrict__ bias, const float* __restrict__ g,
                 const float* __restrict__ be, float* __restrict__ out) {
    __shared__ float s1[256], s2[256];
    const int row = blockIdx.x;
    float lv[6];
#pragma unroll
    for (int s = 0; s < 6; ++s) lv[s] = locs[row * 6 + s];

    float y[3];
    float sum = 0.f, sq = 0.f;
#pragma unroll
    for (int j = 0; j < 3; ++j) {
        const int c = threadIdx.x + j * 256;
        float a = bias[c];
#pragma unroll
        for (int s = 0; s < 6; ++s) a += lv[s] * W[s * 768 + c];
        y[j] = a; sum += a; sq += a * a;
    }
    block_sum2(sum, sq, s1, s2);
    const float m  = sum * (1.0f / 768.0f);
    const float va = sq * (1.0f / 768.0f) - m * m;
    const float rs = rsqrtf(va + 1e-5f);
#pragma unroll
    for (int j = 0; j < 3; ++j) {
        const int c = threadIdx.x + j * 256;
        out[(long long)row * 768 + c] = (y[j] - m) * rs * g[c] + be[c];
    }
}

// ---------------------------------------------------------------------------
// Per-batch max pairwise distance over L*L pairs.
// ---------------------------------------------------------------------------
__global__ __launch_bounds__(256)
void maxd_kernel(const float* __restrict__ locs, float* __restrict__ maxd, int L) {
    __shared__ float cx[512], cy[512], cz[512];
    __shared__ float sr[256];
    const int b = blockIdx.x;
    for (int j = threadIdx.x; j < L; j += 256) {
        cx[j] = locs[((long long)b * L + j) * 6 + 0];
        cy[j] = locs[((long long)b * L + j) * 6 + 1];
        cz[j] = locs[((long long)b * L + j) * 6 + 2];
    }
    __syncthreads();
    float mx = 0.f;
    for (int p = threadIdx.x; p < L * L; p += 256) {
        const int l = p / L, t = p - l * L;
        const float dx = cx[l] - cx[t], dy = cy[l] - cy[t], dz = cz[l] - cz[t];
        mx = fmaxf(mx, sqrtf(dx * dx + dy * dy + dz * dz + 1e-10f));
    }
    mx = block_max(mx, sr);
    if (threadIdx.x == 0) maxd[b] = mx;
}

// ---------------------------------------------------------------------------
// Fused mask + pairwise-loc features + softmax.
// One block per (b, h, l) row of length L=512 (2 elements / thread).
// probs = softmax( log(clip(relu(feat . w + b), 1e-6)) + scale*score , masked )
// ---------------------------------------------------------------------------
__global__ __launch_bounds__(256)
void softmax_loc_kernel(const float* __restrict__ scores,
                        bf16_t* __restrict__ probs,
                        const float* __restrict__ locs,
                        const int* __restrict__ masks,
                        const float* __restrict__ maxd,
                        const float* __restrict__ lw,   // [5][H], layer slice
                        const float* __restrict__ lb,   // [H],    layer slice
                        int Hh, int L) {
    __shared__ float sr[256];
    const int rid = blockIdx.x;
    const int l   = rid % L;
    const int h   = (rid / L) % Hh;
    const int b   = rid / (L * Hh);
    const float* srow = scores + (long long)rid * L;

    const float clx = locs[((long long)b * L + l) * 6 + 0];
    const float cly = locs[((long long)b * L + l) * 6 + 1];
    const float clz = locs[((long long)b * L + l) * 6 + 2];
    const float inv_md = 1.0f / maxd[b];
    const float w0 = lw[0 * Hh + h], w1 = lw[1 * Hh + h], w2 = lw[2 * Hh + h];
    const float w3 = lw[3 * Hh + h], w4 = lw[4 * Hh + h];
    const float lbh = lb[h];

    float vals[2];
#pragma unroll
    for (int e = 0; e < 2; ++e) {
        const int t = threadIdx.x + e * 256;
        const float ctx = locs[((long long)b * L + t) * 6 + 0];
        const float cty = locs[((long long)b * L + t) * 6 + 1];
        const float ctz = locs[((long long)b * L + t) * 6 + 2];
        const float rx = clx - ctx, ry = cly - cty, rz = clz - ctz;
        const float d  = sqrtf(rx * rx + ry * ry + rz * rz + 1e-10f);
        const float d2 = sqrtf(rx * rx + ry * ry + 1e-10f);
        const float inv_d = 1.0f / d, inv_d2 = 1.0f / d2;
        float la = (d * inv_md) * w0 + (rz * inv_d) * w1 + (d2 * inv_d) * w2 +
                   (ry * inv_d2) * w3 + (rx * inv_d2) * w4 + lbh;
        la = fmaxf(la, 0.0f);
        float at = srow[t] * 0.125f;              // scale = 1/sqrt(64)
        const bool valid = masks[(long long)b * L + t] != 0;
        if (!valid) { at = -1e30f; la = 0.0f; }
        vals[e] = logf(fmaxf(la, 1e-6f)) + at;
    }
    const float mx = block_max(fmaxf(vals[0], vals[1]), sr);
    float e0 = expf(vals[0] - mx), e1 = expf(vals[1] - mx);
    const float sm = block_sum(e0 + e1, sr);
    const float inv = 1.0f / sm;
    probs[(long long)rid * L + threadIdx.x]       = (bf16_t)(e0 * inv);
    probs[(long long)rid * L + threadIdx.x + 256] = (bf16_t)(e1 * inv);
}

// ---------------------------------------------------------------------------
// x_out = LN1( x + LN0(ao + x) ), one block per row of D=768.
// ---------------------------------------------------------------------------
__global__ __launch_bounds__(256)
void ln_double_kernel(const float* __restrict__ ao, const float* __restrict__ x_in,
                      const float* __restrict__ g0, const float* __restrict__ b0,
                      const float* __restrict__ g1, const float* __restrict__ b1,
                      float* __restrict__ x_out, bf16_t* __restrict__ xb_out) {
    __shared__ float s1[256], s2[256];
    const long long row = blockIdx.x;
    float xr[3], t[3];
    float sum = 0.f, sq = 0.f;
#pragma unroll
    for (int j = 0; j < 3; ++j) {
        const int c = threadIdx.x + j * 256;
        xr[j] = x_in[row * 768 + c];
        t[j]  = ao[row * 768 + c] + xr[j];
        sum += t[j]; sq += t[j] * t[j];
    }
    block_sum2(sum, sq, s1, s2);
    float m  = sum * (1.0f / 768.0f);
    float va = sq * (1.0f / 768.0f) - m * m;
    float rs = rsqrtf(va + 1e-5f);
    float u[3];
    sum = 0.f; sq = 0.f;
#pragma unroll
    for (int j = 0; j < 3; ++j) {
        const int c = threadIdx.x + j * 256;
        const float x2 = (t[j] - m) * rs * g0[c] + b0[c];
        u[j] = xr[j] + x2;
        sum += u[j]; sq += u[j] * u[j];
    }
    block_sum2(sum, sq, s1, s2);
    m  = sum * (1.0f / 768.0f);
    va = sq * (1.0f / 768.0f) - m * m;
    rs = rsqrtf(va + 1e-5f);
#pragma unroll
    for (int j = 0; j < 3; ++j) {
        const int c = threadIdx.x + j * 256;
        const float o = (u[j] - m) * rs * g1[c] + b1[c];
        x_out[row * 768 + c]  = o;
        xb_out[row * 768 + c] = (bf16_t)o;
    }
}

// ---------------------------------------------------------------------------
// dst = LN( x + ff ), one block per row of D=768.  xb optional.
// ---------------------------------------------------------------------------
__global__ __launch_bounds__(256)
void ln_single_kernel(const float* __restrict__ ff, const float* __restrict__ x_in,
                      const float* __restrict__ g, const float* __restrict__ be,
                      float* __restrict__ dst, bf16_t* __restrict__ xb_out) {
    __shared__ float s1[256], s2[256];
    const long long row = blockIdx.x;
    float t[3];
    float sum = 0.f, sq = 0.f;
#pragma unroll
    for (int j = 0; j < 3; ++j) {
        const int c = threadIdx.x + j * 256;
        t[j] = x_in[row * 768 + c] + ff[row * 768 + c];
        sum += t[j]; sq += t[j] * t[j];
    }
    block_sum2(sum, sq, s1, s2);
    const float m  = sum * (1.0f / 768.0f);
    const float va = sq * (1.0f / 768.0f) - m * m;
    const float rs = rsqrtf(va + 1e-5f);
#pragma unroll
    for (int j = 0; j < 3; ++j) {
        const int c = threadIdx.x + j * 256;
        const float o = (t[j] - m) * rs * g[c] + be[c];
        dst[row * 768 + c] = o;
        if (xb_out) xb_out[row * 768 + c] = (bf16_t)o;
    }
}

// ---------------------------------------------------------------------------
// Host side
// ---------------------------------------------------------------------------
static inline void launch_gemm(hipStream_t stream,
                               const bf16_t* A, const bf16_t* B,
                               float* C, bf16_t* Cb, const float* bias,
                               int M, int N, int K, int lda, int ldb, int ldc,
                               int batches, int bat_h,
                               long long sAb, long long sAh,
                               long long sBb, long long sBh,
                               long long sCb, long long sCh,
                               int cb_vt, int act) {
    dim3 grid(N / 64, M / 128, batches);
    gemm_bf16_wmma_kernel<<<grid, 256, 0, stream>>>(
        A, B, C, Cb, bias, M, N, K, lda, ldb, ldc, bat_h,
        sAb, sAh, sBb, sBh, sCb, sCh, cb_vt, act);
}

extern "C" void kernel_launch(void* const* d_in, const int* in_sizes, int n_in,
                              void* d_out, int out_size, void* d_ws, size_t ws_size,
                              hipStream_t stream) {
    (void)in_sizes; (void)n_in; (void)out_size; (void)ws_size;
    constexpr int B = 16, L = 512, D = 768, H = 12, DK = 64, FF = 2048;
    constexpr long long ND = (long long)B * L * D;        // 6291456
    constexpr long long NS = (long long)B * H * L * L;    // 50331648

    // ---- inputs (setup_inputs order) ----
    const float* obj_embeds = (const float*)d_in[0];
    const float* obj_locs   = (const float*)d_in[1];
    const int*   obj_masks  = (const int*)d_in[2];
    const float* loc_w      = (const float*)d_in[3];
    const float* loc_b      = (const float*)d_in[4];
    const float* loc_ln_g   = (const float*)d_in[5];
    const float* loc_ln_b   = (const float*)d_in[6];
    const float* Wq  = (const float*)d_in[7];   const float* bq  = (const float*)d_in[8];
    const float* Wk  = (const float*)d_in[9];   const float* bk  = (const float*)d_in[10];
    const float* Wv  = (const float*)d_in[11];  const float* bv  = (const float*)d_in[12];
    const float* Wfc = (const float*)d_in[13];  const float* bfc = (const float*)d_in[14];
    const float* loc_fc_w = (const float*)d_in[15];
    const float* loc_fc_b = (const float*)d_in[16];
    const float* ln0_g = (const float*)d_in[17]; const float* ln0_b = (const float*)d_in[18];
    const float* W1 = (const float*)d_in[19];   const float* b1 = (const float*)d_in[20];
    const float* W2 = (const float*)d_in[21];   const float* b2 = (const float*)d_in[22];
    const float* ln1_g = (const float*)d_in[23]; const float* ln1_b = (const float*)d_in[24];
    const float* ln2_g = (const float*)d_in[25]; const float* ln2_b = (const float*)d_in[26];

    // ---- workspace bump allocator ----
    char* ws = (char*)d_ws;
    size_t off = 0;
    auto alloc = [&](size_t bytes) -> void* {
        void* p = ws + off;
        off = (off + bytes + 255) & ~(size_t)255;
        return p;
    };
    float*  x    = (float*)alloc(ND * 4);
    bf16_t* xb   = (bf16_t*)alloc(ND * 2);
    float*  qpos = (float*)alloc(ND * 4);
    float*  maxd = (float*)alloc(256);
    // all weights pre-transposed to [N][K] bf16
    bf16_t* wq_t  = (bf16_t*)alloc(4ll * D * D * 2);
    bf16_t* wk_t  = (bf16_t*)alloc(4ll * D * D * 2);
    bf16_t* wv_t  = (bf16_t*)alloc(4ll * D * D * 2);
    bf16_t* wfc_t = (bf16_t*)alloc(4ll * D * D * 2);
    bf16_t* w1_t  = (bf16_t*)alloc(4ll * D * FF * 2);
    bf16_t* w2_t  = (bf16_t*)alloc(4ll * FF * D * 2);
    bf16_t* q  = (bf16_t*)alloc(ND * 2);
    bf16_t* k  = (bf16_t*)alloc(ND * 2);
    bf16_t* vT = (bf16_t*)alloc(ND * 2);     // (B, H, DK, L)
    float*  scores = (float*)alloc(NS * 4);
    bf16_t* probs  = (bf16_t*)alloc(NS * 2);
    bf16_t* ao     = (bf16_t*)alloc(ND * 2);
    float*  attn_out = (float*)alloc(ND * 4);
    // FFN scratch aliases the (dead at that point) scores buffer
    bf16_t* h1 = (bf16_t*)scores;
    float*  ff = (float*)((char*)scores +
                          (((long long)B * L * FF * 2 + 255) & ~255ll));

    // ---- weight transpose+convert to bf16 [N][K] ----
    {
        dim3 gDD(D / 32, D / 32, 4);
        transpose_bf16_kernel<<<gDD, 256, 0, stream>>>(Wq,  wq_t,  D, D);
        transpose_bf16_kernel<<<gDD, 256, 0, stream>>>(Wk,  wk_t,  D, D);
        transpose_bf16_kernel<<<gDD, 256, 0, stream>>>(Wv,  wv_t,  D, D);
        transpose_bf16_kernel<<<gDD, 256, 0, stream>>>(Wfc, wfc_t, D, D);
        dim3 g1(FF / 32, D / 32, 4);   // W1: K=D, N=FF
        transpose_bf16_kernel<<<g1, 256, 0, stream>>>(W1, w1_t, D, FF);
        dim3 g2(D / 32, FF / 32, 4);   // W2: K=FF, N=D
        transpose_bf16_kernel<<<g2, 256, 0, stream>>>(W2, w2_t, FF, D);
    }

    // ---- query_pos and per-batch max distance ----
    qpos_kernel<<<B * L, 256, 0, stream>>>(obj_locs, loc_w, loc_b,
                                           loc_ln_g, loc_ln_b, qpos);
    maxd_kernel<<<B, 256, 0, stream>>>(obj_locs, maxd, L);

    const int M = B * L;                 // 8192
    for (int i = 0; i < 4; ++i) {
        const long long wD = (long long)i * D * D;
        const long long wF = (long long)i * D * FF;

        // x = (prev x or obj_embeds) + query_pos
        addpos_kernel<<<2048, 256, 0, stream>>>(i == 0 ? obj_embeds : x,
                                                qpos, x, xb, ND);

        // Q, K projections (bf16 out), V projection stored transposed (B,H,DK,L)
        launch_gemm(stream, xb, wq_t + wD, nullptr, q, bq + i * D,
                    M, D, D, D, D, D, 1, 1, 0, 0, 0, 0, 0, 0, 0, 0);
        launch_gemm(stream, xb, wk_t + wD, nullptr, k, bk + i * D,
                    M, D, D, D, D, D, 1, 1, 0, 0, 0, 0, 0, 0, 0, 0);
        launch_gemm(stream, xb, wv_t + wD, nullptr, vT, bv + i * D,
                    M, D, D, D, D, D, 1, 1, 0, 0, 0, 0, 0, 0, /*cb_vt*/1, 0);

        // scores[b,h] = q[b,:,h,:] @ k[b,:,h,:]^T   (k rows are N-major natively)
        launch_gemm(stream, q, k, scores, nullptr, nullptr,
                    L, L, DK, D, D, L, B * H, H,
                    (long long)L * D, DK,                    // A per-b, per-h
                    (long long)L * D, DK,                    // B per-b, per-h
                    (long long)H * L * L, (long long)L * L,  // C per-b, per-h
                    0, 0);

        // fused mask + loc-feature + softmax -> probs (bf16)
        softmax_loc_kernel<<<B * H * L, 256, 0, stream>>>(
            scores, probs, obj_locs, obj_masks, maxd,
            loc_fc_w + (long long)i * 5 * H, loc_fc_b + (long long)i * H, H, L);

        // head out[b,:,h,:] = probs[b,h] @ v[b,:,h,:]  (B = vT, N-major)
        launch_gemm(stream, probs, vT, nullptr, ao, nullptr,
                    L, DK, L, L, L, D, B * H, H,
                    (long long)H * L * L, (long long)L * L,        // A (probs)
                    (long long)H * DK * L, (long long)DK * L,      // B (vT)
                    (long long)L * D, DK,                          // C (ao)
                    0, 0);

        // output projection (f32)
        launch_gemm(stream, ao, wfc_t + wD, attn_out, nullptr, bfc + i * D,
                    M, D, D, D, D, D, 1, 1, 0, 0, 0, 0, 0, 0, 0, 0);

        // x = LN1( x + LN0(attn_out + x) )
        ln_double_kernel<<<M, 256, 0, stream>>>(attn_out, x,
                                                ln0_g + i * D, ln0_b + i * D,
                                                ln1_g + i * D, ln1_b + i * D,
                                                x, xb);

        // FFN: h1 = gelu(x @ W1 + b1) (bf16);  ff = h1 @ W2 + b2 (f32)
        launch_gemm(stream, xb, w1_t + wF, nullptr, h1, b1 + i * FF,
                    M, FF, D, D, D, FF, 1, 1, 0, 0, 0, 0, 0, 0, 0, /*gelu*/1);
        launch_gemm(stream, h1, w2_t + wF, ff, nullptr, b2 + i * D,
                    M, D, FF, FF, FF, D, 1, 1, 0, 0, 0, 0, 0, 0, 0, 0);

        // x = LN2( x + ff );  final layer writes straight to d_out
        ln_single_kernel<<<M, 256, 0, stream>>>(
            ff, x, ln2_g + i * D, ln2_b + i * D,
            (i == 3) ? (float*)d_out : x,
            (i == 3) ? nullptr : xb);
    }
}
